// Llama4TextDecoderLayer_37589553774976
// MI455X (gfx1250) — matrix operand order repaired and checked by reading, not compiled, not run on hardware
//
#include <hip/hip_runtime.h>
#include <hip/hip_bf16.h>

// ---------------------------------------------------------------------------
// Types for CDNA5 WMMA (wave32): v_wmma_f32_16x16x32_bf16
// ---------------------------------------------------------------------------
typedef __attribute__((ext_vector_type(16))) __bf16 bf16x16;
typedef __attribute__((ext_vector_type(8)))  float  f32x8;

union FragU { uint4 u[2]; bf16x16 v; };

// ---------------------------------------------------------------------------
// gfx1250 async global->LDS copy (ASYNCcnt-tracked), guarded by __has_builtin
// Builtin prototype (from clang diagnostic): (v4i AS1*, v4i AS3*, Imm, Imm)
// ---------------------------------------------------------------------------
#if defined(__has_builtin)
#if __has_builtin(__builtin_amdgcn_global_load_async_to_lds_b128) && \
    __has_builtin(__builtin_amdgcn_s_wait_asynccnt)
#define HAVE_ASYNC_LDS 1
#endif
#endif
#ifndef HAVE_ASYNC_LDS
#define HAVE_ASYNC_LDS 0
#endif

#if HAVE_ASYNC_LDS
typedef int v4i __attribute__((vector_size(16)));
typedef __attribute__((address_space(1))) v4i as1_v4i;
typedef __attribute__((address_space(3))) v4i as3_v4i;

__device__ __forceinline__ void async_copy16(const void* gsrc, void* ldst) {
    // ISA 10.2: for the LDS aperture, flat addr[31:0] is the LDS offset, so an
    // integer round-trip yields valid AS1/AS3 pointers without addrspacecast.
    as1_v4i* g = (as1_v4i*)(unsigned long long)gsrc;
    as3_v4i* l = (as3_v4i*)(unsigned int)(unsigned long long)ldst;
    __builtin_amdgcn_global_load_async_to_lds_b128(g, l, 0, 0);
}
__device__ __forceinline__ void async_wait0() {
    __builtin_amdgcn_s_wait_asynccnt(0);
}
#endif

// A-matrix 16x32 bf16 fragment (ISA 7.12.2): lane r=lane&15 holds row r;
// lane-half kh=lane>>4 selects K sets {kh*8..kh*8+7, 16+kh*8..16+kh*8+7}.
__device__ __forceinline__ bf16x16 load_a_frag(const __bf16* rowPtr, int kh) {
    FragU f;
    f.u[0] = *(const uint4*)(rowPtr + kh * 8);
    f.u[1] = *(const uint4*)(rowPtr + 16 + kh * 8);
    return f.v;
}

// B-matrix 32x16 bf16 fragment: lane r holds output column r; lane-half kh
// selects K sets {kh*8..kh*8+7, 16+kh*8..16+kh*8+7}. colPtr points at the
// contiguous-K storage for this lane's column.
__device__ __forceinline__ bf16x16 load_b_frag(const __bf16* colPtr, int kh) {
    FragU f;
    f.u[0] = *(const uint4*)(colPtr + kh * 8);
    f.u[1] = *(const uint4*)(colPtr + 16 + kh * 8);
    return f.v;
}

__device__ __forceinline__ f32x8 wmma_bf16(bf16x16 a, bf16x16 b, f32x8 c) {
    return __builtin_amdgcn_wmma_f32_16x16x32_bf16(
        false, a, false, b, (short)0, c, false, false);
}

__device__ __forceinline__ f32x8 zero8() {
    f32x8 z = {0.f, 0.f, 0.f, 0.f, 0.f, 0.f, 0.f, 0.f};
    return z;
}

// ---------------------------------------------------------------------------
// RMSNorm: fp32 row (4096) -> bf16 row, scaled by weight
// ---------------------------------------------------------------------------
__global__ __launch_bounds__(256) void rmsnorm_kernel(
    const float* __restrict__ X, const float* __restrict__ W,
    __bf16* __restrict__ Out) {
    __shared__ float red[8];
    const int s = blockIdx.x, tid = threadIdx.x;
    const float* xp = X + (size_t)s * 4096;
    float v[16];
    float ss = 0.f;
#pragma unroll
    for (int i = 0; i < 4; i++) {
        float4 f = ((const float4*)xp)[tid * 4 + i];
        v[i * 4 + 0] = f.x; v[i * 4 + 1] = f.y;
        v[i * 4 + 2] = f.z; v[i * 4 + 3] = f.w;
        ss += f.x * f.x + f.y * f.y + f.z * f.z + f.w * f.w;
    }
#pragma unroll
    for (int d = 1; d < 32; d <<= 1) ss += __shfl_xor(ss, d, 32);
    if ((tid & 31) == 0) red[tid >> 5] = ss;
    __syncthreads();
    float tot = 0.f;
#pragma unroll
    for (int i = 0; i < 8; i++) tot += red[i];
    const float sc = rsqrtf(tot * (1.0f / 4096.0f) + 1e-5f);
    __bf16* op = Out + (size_t)s * 4096 + tid * 16;
    const float* wp = W + tid * 16;
#pragma unroll
    for (int i = 0; i < 16; i++) op[i] = (__bf16)(v[i] * sc * wp[i]);
}

// ---------------------------------------------------------------------------
// GEMM: C[M,N] (+Res) = A_bf16[M,K] * B_f32[K,N]   (weights converted in-flight)
// Block 256 thr = 8 waves; tile 128x128x32; wave tile 64x32 = 4x2 WMMA accs.
// A tile staged via async global->LDS DMA; B tile converted fp32->bf16 and
// stored transposed [n][k] with packed dword stores.
// ---------------------------------------------------------------------------
#define BK   32
#define LDSK 40   // padded stride -> conflict-free uint4 fragment loads

__global__ __launch_bounds__(256) void gemm_bf16_kernel(
    const __bf16* __restrict__ A, const float* __restrict__ Bm,
    const float* __restrict__ Res, float* __restrict__ C,
    int M, int N, int K) {
    __shared__ __bf16 Alds[128 * LDSK];
    __shared__ __bf16 Blds[128 * LDSK];   // stored transposed: [n][k]

    const int tid  = threadIdx.x;
    const int m0   = blockIdx.y * 128;
    const int n0   = blockIdx.x * 128;
    const int lane = tid & 31;
    const int wid  = tid >> 5;
    const int wm   = (wid >> 2) * 64;     // 0 / 64
    const int wn   = (wid & 3) * 32;      // 0 / 32 / 64 / 96
    const int r    = lane & 15;
    const int kh   = lane >> 4;

    f32x8 acc[4][2];
#pragma unroll
    for (int i = 0; i < 4; i++)
#pragma unroll
        for (int j = 0; j < 2; j++) acc[i][j] = zero8();

    const int ar = tid >> 1, aq = tid & 1;           // A: 2 thr/row, 16 bf16 each
    const int kp = tid >> 4, ncol = (tid & 15) * 8;  // B: thread owns k-pair x 8 cols

    for (int k0 = 0; k0 < K; k0 += BK) {
        // ---- stage A tile (bf16, straight copy, async DMA to LDS) ----
        const __bf16* asrc = A + (size_t)(m0 + ar) * K + k0 + aq * 16;
        __bf16*       adst = &Alds[ar * LDSK + aq * 16];
#if HAVE_ASYNC_LDS
        async_copy16(asrc, adst);
        async_copy16(asrc + 8, adst + 8);
#else
        uint4 a0 = ((const uint4*)asrc)[0];
        uint4 a1 = ((const uint4*)asrc)[1];
        *(uint4*)(adst)     = a0;
        *(uint4*)(adst + 8) = a1;
#endif
        // ---- stage B tile (fp32 -> bf16, transposed to [n][k], packed b32) ----
        const float* bsrc0 = Bm + (size_t)(k0 + 2 * kp) * N + n0 + ncol;
        const float* bsrc1 = bsrc0 + N;
        __builtin_prefetch(bsrc0 + (size_t)BK * N, 0, 0);   // speculative, drop-safe
        float4 r00 = ((const float4*)bsrc0)[0];
        float4 r01 = ((const float4*)bsrc0)[1];
        float4 r10 = ((const float4*)bsrc1)[0];
        float4 r11 = ((const float4*)bsrc1)[1];
        float lo[8] = {r00.x, r00.y, r00.z, r00.w, r01.x, r01.y, r01.z, r01.w};
        float hi[8] = {r10.x, r10.y, r10.z, r10.w, r11.x, r11.y, r11.z, r11.w};
#pragma unroll
        for (int i = 0; i < 8; i++) {
            union { __bf16 h[2]; unsigned int u; } pk;
            pk.h[0] = (__bf16)lo[i];
            pk.h[1] = (__bf16)hi[i];
            *(unsigned int*)&Blds[(ncol + i) * LDSK + 2 * kp] = pk.u;
        }
#if HAVE_ASYNC_LDS
        async_wait0();
#endif
        __syncthreads();

        bf16x16 af[4], bfr[2];
#pragma unroll
        for (int i = 0; i < 4; i++)
            af[i] = load_a_frag(&Alds[(wm + i * 16 + r) * LDSK], kh);
#pragma unroll
        for (int j = 0; j < 2; j++)
            bfr[j] = load_b_frag(&Blds[(wn + j * 16 + r) * LDSK], kh);
#pragma unroll
        for (int i = 0; i < 4; i++)
#pragma unroll
            for (int j = 0; j < 2; j++)
                acc[i][j] = wmma_bf16(af[i], bfr[j], acc[i][j]);
        __syncthreads();
    }

    // ---- epilogue: C/D layout -> row = base + e + 8*kh, col = base + r ----
    // Single uniform Res branch; row-pointer walk so stores become base+imm.
    const size_t cbase = (size_t)(m0 + wm + 8 * kh) * N + n0 + wn + r;
    if (Res) {
#pragma unroll
        for (int i = 0; i < 4; i++) {
            float*       crow = C   + cbase + (size_t)(i * 16) * N;
            const float* rrow = Res + cbase + (size_t)(i * 16) * N;
#pragma unroll
            for (int e = 0; e < 8; e++) {
                crow[0]  = acc[i][0][e] + rrow[0];
                crow[16] = acc[i][1][e] + rrow[16];
                crow += N; rrow += N;
            }
        }
    } else {
#pragma unroll
        for (int i = 0; i < 4; i++) {
            float* crow = C + cbase + (size_t)(i * 16) * N;
#pragma unroll
            for (int e = 0; e < 8; e++) {
                crow[0]  = acc[i][0][e];
                crow[16] = acc[i][1][e];
                crow += N;
            }
        }
    }
}

// ---------------------------------------------------------------------------
// RoPE + L2 norm: fp32 (S, nheads, 128) -> bf16; 8 threads per head row
// ---------------------------------------------------------------------------
__global__ void rope_l2_kernel(const float* __restrict__ X,
                               const float* __restrict__ Cos,
                               const float* __restrict__ Sin,
                               __bf16* __restrict__ Out, int nheads) {
    const int s   = blockIdx.x;
    const int tid = threadIdx.x;          // nheads * 8 threads
    const int hh  = tid >> 3, sub = tid & 7;
    const float* xp = X + ((size_t)s * nheads + hh) * 128;
    __bf16*      op = Out + ((size_t)s * nheads + hh) * 128;
    const float* cp = Cos + (size_t)s * 64;
    const float* sp = Sin + (size_t)s * 64;
    float orv[8], oiv[8];
    float ss = 0.f;
#pragma unroll
    for (int i = 0; i < 8; i++) {
        const int p = sub * 8 + i;
        const float xr = xp[2 * p], xi = xp[2 * p + 1];
        const float c = cp[p], sn = sp[p];
        const float a = xr * c - xi * sn;
        const float b = xr * sn + xi * c;
        orv[i] = a; oiv[i] = b;
        ss += a * a + b * b;
    }
    ss += __shfl_xor(ss, 1, 32);
    ss += __shfl_xor(ss, 2, 32);
    ss += __shfl_xor(ss, 4, 32);
    const float sc = rsqrtf(ss * (1.0f / 128.0f) + 1e-5f);
#pragma unroll
    for (int i = 0; i < 8; i++) {
        const int p = sub * 8 + i;
        op[2 * p]     = (__bf16)(orv[i] * sc);
        op[2 * p + 1] = (__bf16)(oiv[i] * sc);
    }
}

// ---------------------------------------------------------------------------
// V transpose: f32 (S, 8*128) -> bf16 [kv][d][s]  (contiguous-K for P*V frags)
// ---------------------------------------------------------------------------
__global__ void vtrans_kernel(const float* __restrict__ V,
                              __bf16* __restrict__ Vt) {
    const size_t i = (size_t)blockIdx.x * blockDim.x + threadIdx.x;
    if (i >= (size_t)2048 * 1024) return;
    const int sidx  = (int)(i & 2047);
    const int dglob = (int)(i >> 11);
    Vt[i] = (__bf16)V[(size_t)sidx * 1024 + dglob];
}

// ---------------------------------------------------------------------------
// Flash attention, one wave per (16-row q block, head). Causal, online softmax.
// Q: bf16 (S,32,128)  K: bf16 (S,8,128)  Vt: bf16 [8][128][2048]
// ---------------------------------------------------------------------------
__global__ __launch_bounds__(32) void flash_attn_kernel(
    const __bf16* __restrict__ Q, const __bf16* __restrict__ Km,
    const __bf16* __restrict__ Vt, __bf16* __restrict__ O) {
    __shared__ __bf16 P[16 * LDSK];
    const int lane = threadIdx.x;
    const int r = lane & 15, kh = lane >> 4;
    const int s0 = blockIdx.x * 16;
    const int h  = blockIdx.y;
    const int kv = h >> 2;                 // 32 heads / 8 kv heads
    const float scale = 0.08838834764831845f;   // 128^-0.5

    // resident Q fragments: 4 chunks of K=32 over HD=128
    bf16x16 qf[4];
    const __bf16* qrow = Q + (size_t)(s0 + r) * 4096 + h * 128;
#pragma unroll
    for (int c = 0; c < 4; c++) qf[c] = load_a_frag(qrow + c * 32, kh);

    f32x8 oacc[8];
#pragma unroll
    for (int t = 0; t < 8; t++) oacc[t] = zero8();
    float mi[8], li[8];
#pragma unroll
    for (int e = 0; e < 8; e++) { mi[e] = -3e38f; li[e] = 0.f; }

    const int nkb = (s0 + 16 + 31) / 32;   // causal: keys <= s0+15
    for (int kb = 0; kb < nkb; ++kb) {
        const int kbase = kb * 32;
        // ---- S = Q * K^T (two 16x16 tiles covering 32 keys) ----
        f32x8 sv0 = zero8(), sv1 = zero8();
#pragma unroll
        for (int c = 0; c < 4; c++) {
            const __bf16* k0p = Km + (size_t)(kbase + r) * 1024 + kv * 128 + c * 32;
            const __bf16* k1p = Km + (size_t)(kbase + 16 + r) * 1024 + kv * 128 + c * 32;
            sv0 = wmma_bf16(qf[c], load_b_frag(k0p, kh), sv0);
            sv1 = wmma_bf16(qf[c], load_b_frag(k1p, kh), sv1);
        }
        // ---- online softmax (row reductions across 16 lanes) ----
        float alpha[8];
#pragma unroll
        for (int e = 0; e < 8; e++) {
            const int rowg = s0 + e + 8 * kh;
            float x0 = sv0[e] * scale;
            float x1 = sv1[e] * scale;
            if (kbase + r      > rowg) x0 = -1e9f;   // causal mask (== NEG)
            if (kbase + 16 + r > rowg) x1 = -1e9f;
            float m = fmaxf(x0, x1);
            m = fmaxf(m, __shfl_xor(m, 1, 32));
            m = fmaxf(m, __shfl_xor(m, 2, 32));
            m = fmaxf(m, __shfl_xor(m, 4, 32));
            m = fmaxf(m, __shfl_xor(m, 8, 32));
            const float mnew = fmaxf(mi[e], m);
            const float a  = __expf(mi[e] - mnew);
            const float p0 = __expf(x0 - mnew);
            const float p1 = __expf(x1 - mnew);
            float rs = p0 + p1;
            rs += __shfl_xor(rs, 1, 32);
            rs += __shfl_xor(rs, 2, 32);
            rs += __shfl_xor(rs, 4, 32);
            rs += __shfl_xor(rs, 8, 32);
            li[e] = li[e] * a + rs;
            mi[e] = mnew;
            alpha[e] = a;
            P[(e + 8 * kh) * LDSK + r]      = (__bf16)p0;
            P[(e + 8 * kh) * LDSK + 16 + r] = (__bf16)p1;
        }
        __syncthreads();
        // ---- O = diag(alpha)*O + P * V ----
        const bf16x16 pf = load_a_frag(&P[r * LDSK], kh);
#pragma unroll
        for (int t = 0; t < 8; t++) {
#pragma unroll
            for (int e = 0; e < 8; e++) oacc[t][e] *= alpha[e];
            const __bf16* vcol =
                Vt + (size_t)kv * (128 * 2048) + (size_t)(t * 16 + r) * 2048 + kbase;
            oacc[t] = wmma_bf16(pf, load_b_frag(vcol, kh), oacc[t]);
        }
        __syncthreads();
    }
    // ---- normalize + store bf16 (S, 32*128) ----
#pragma unroll
    for (int e = 0; e < 8; e++) {
        const float inv = li[e] > 0.f ? 1.0f / li[e] : 0.f;
        __bf16* orow = O + (size_t)(s0 + e + 8 * kh) * 4096 + h * 128;
#pragma unroll
        for (int t = 0; t < 8; t++)
            orow[t * 16 + r] = (__bf16)(oacc[t][e] * inv);
    }
}

// ---------------------------------------------------------------------------
// SiLU(gate) * up -> bf16
// ---------------------------------------------------------------------------
__global__ void silu_mul_kernel(const float* __restrict__ G,
                                const float* __restrict__ U,
                                __bf16* __restrict__ T, size_t n) {
    const size_t i = (size_t)blockIdx.x * blockDim.x + threadIdx.x;
    if (i < n) {
        const float g = G[i];
        const float sg = g / (1.f + __expf(-g));
        T[i] = (__bf16)(sg * U[i]);
    }
}

// ---------------------------------------------------------------------------
// Orchestration
// ---------------------------------------------------------------------------
extern "C" void kernel_launch(void* const* d_in, const int* in_sizes, int n_in,
                              void* d_out, int out_size, void* d_ws, size_t ws_size,
                              hipStream_t stream) {
    (void)in_sizes; (void)n_in; (void)out_size; (void)ws_size;
    constexpr size_t S_ = 2048, D_ = 4096, DKV = 1024, DFF_ = 16384;

    const float* hidden = (const float*)d_in[0];
    /* d_in[1] attn_mask: causal mask applied analytically in flash kernel */
    const float* cosb   = (const float*)d_in[2];
    const float* sinb   = (const float*)d_in[3];
    const float* ln1    = (const float*)d_in[4];
    const float* wq     = (const float*)d_in[5];
    const float* wk     = (const float*)d_in[6];
    const float* wv     = (const float*)d_in[7];
    const float* wo     = (const float*)d_in[8];
    const float* ln2    = (const float*)d_in[9];
    const float* gate_w = (const float*)d_in[10];
    const float* up_w   = (const float*)d_in[11];
    const float* down_w = (const float*)d_in[12];
    float* out = (float*)d_out;

    char* ws = (char*)d_ws;
    size_t off = 0;
    auto alloc = [&](size_t bytes) -> void* {
        void* p = ws + off;
        off += (bytes + 255) & ~(size_t)255;
        return p;
    };
    __bf16* xn    = (__bf16*)alloc(S_ * D_ * 2);
    float*  qf32  = (float*) alloc(S_ * D_ * 4);
    float*  kf32  = (float*) alloc(S_ * DKV * 4);
    float*  vf32  = (float*) alloc(S_ * DKV * 4);
    __bf16* qb16  = (__bf16*)alloc(S_ * D_ * 2);
    __bf16* kb16  = (__bf16*)alloc(S_ * DKV * 2);
    __bf16* vt16  = (__bf16*)alloc(S_ * DKV * 2);
    __bf16* attnb = (__bf16*)alloc(S_ * D_ * 2);
    float*  hbuf  = (float*) alloc(S_ * D_ * 4);
    __bf16* x2    = (__bf16*)alloc(S_ * D_ * 2);
    float*  gatef = (float*) alloc(S_ * DFF_ * 4);
    float*  upf   = (float*) alloc(S_ * DFF_ * 4);
    __bf16* tb16  = (__bf16*)alloc(S_ * DFF_ * 2);

    // 1) ln1
    rmsnorm_kernel<<<S_, 256, 0, stream>>>(hidden, ln1, xn);
    // 2) q/k/v projections (bf16 x fp32-weights -> f32)
    gemm_bf16_kernel<<<dim3(D_ / 128, S_ / 128), 256, 0, stream>>>(
        xn, wq, nullptr, qf32, (int)S_, (int)D_, (int)D_);
    gemm_bf16_kernel<<<dim3(DKV / 128, S_ / 128), 256, 0, stream>>>(
        xn, wk, nullptr, kf32, (int)S_, (int)DKV, (int)D_);
    gemm_bf16_kernel<<<dim3(DKV / 128, S_ / 128), 256, 0, stream>>>(
        xn, wv, nullptr, vf32, (int)S_, (int)DKV, (int)D_);
    // 3) rope + l2 norm, V transpose
    rope_l2_kernel<<<S_, 256, 0, stream>>>(qf32, cosb, sinb, qb16, 32);
    rope_l2_kernel<<<S_, 64, 0, stream>>>(kf32, cosb, sinb, kb16, 8);
    vtrans_kernel<<<(S_ * DKV) / 256, 256, 0, stream>>>(vf32, vt16);
    // 4) flash attention
    flash_attn_kernel<<<dim3(S_ / 16, 32), 32, 0, stream>>>(qb16, kb16, vt16, attnb);
    // 5) O projection + residual
    gemm_bf16_kernel<<<dim3(D_ / 128, S_ / 128), 256, 0, stream>>>(
        attnb, wo, hidden, hbuf, (int)S_, (int)D_, (int)D_);
    // 6) ln2
    rmsnorm_kernel<<<S_, 256, 0, stream>>>(hbuf, ln2, x2);
    // 7) FFN
    gemm_bf16_kernel<<<dim3(DFF_ / 128, S_ / 128), 256, 0, stream>>>(
        x2, gate_w, nullptr, gatef, (int)S_, (int)DFF_, (int)D_);
    gemm_bf16_kernel<<<dim3(DFF_ / 128, S_ / 128), 256, 0, stream>>>(
        x2, up_w, nullptr, upf, (int)S_, (int)DFF_, (int)D_);
    silu_mul_kernel<<<(S_ * DFF_) / 256, 256, 0, stream>>>(
        gatef, upf, tb16, S_ * DFF_);
    // 8) down projection + residual -> output
    gemm_bf16_kernel<<<dim3(D_ / 128, S_ / 128), 256, 0, stream>>>(
        tb16, down_w, hbuf, out, (int)S_, (int)D_, (int)DFF_);
}
